// ScenePromptModule_V2_20392504721505
// MI455X (gfx1250) — compile-verified, matrix operand order checked by reading, not run on hardware
//
#include <hip/hip_runtime.h>
#include <hip/hip_bf16.h>

typedef __attribute__((ext_vector_type(16))) _Float16 v16h;
typedef __attribute__((ext_vector_type(8)))  float    v8f;

// ---------------------------------------------------------------------------
// WMMA helpers (wave32, V_WMMA_F32_16X16X32_F16).
// A fragment (16x32 f16) from row-major storage: per-lane halves are two
// contiguous 8-half chunks -> lowers to 2x b128 loads.
// B fragments are consumed from a PRE-SWIZZLED buffer where each lane's 16
// halves of each 32x16 K-tile are contiguous -> 1x v16h (2x b128) load.
// ---------------------------------------------------------------------------
__device__ __forceinline__ v16h frag_a(const _Float16* p, int ld, int row0, int k0) {
  int lane = threadIdx.x & 31;
  int m    = lane & 15;
  int kh   = (lane >> 4) << 3;
  const _Float16* r = p + (size_t)(row0 + m) * ld + k0 + kh;
  v16h f;
#pragma unroll
  for (int v = 0; v < 8; ++v) {
    int kb = ((v & 4) << 2) + ((v & 3) << 1);   // v<4: 2v, v>=4: 16+2(v-4)
    f[2 * v]     = r[kb];
    f[2 * v + 1] = r[kb + 1];
  }
  return f;
}

// swizzled B fragment: tile (kt, nt), KT = Kd/32 tiles along K
__device__ __forceinline__ v16h frag_b_sw(const _Float16* p, int KT, int kt, int nt) {
  int lane = threadIdx.x & 31;
  return *(const v16h*)(p + ((((size_t)nt * KT + kt) * 32 + lane) << 4));
}

__device__ __forceinline__ v8f wmma32(v16h a, v16h b, v8f c) {
  return __builtin_amdgcn_wmma_f32_16x16x32_f16(false, a, false, b, (short)0, c,
                                                false, false);
}

// one 16x16 C tile of A(MxKd,row-major,f16) * Bsw(swizzled KdxN f16)
__device__ __forceinline__ v8f gemm_tile(const _Float16* A, const _Float16* Bsw,
                                         int Kd, int row0, int nt) {
  int KT = Kd >> 5;
  v8f acc = {0.f, 0.f, 0.f, 0.f, 0.f, 0.f, 0.f, 0.f};
  for (int kt = 0; kt < KT; ++kt)
    acc = wmma32(frag_a(A, Kd, row0, kt * 32), frag_b_sw(Bsw, KT, kt, nt), acc);
  return acc;
}

// ---------------------------------------------------------------------------
// Prep: build swizzled f16 B from row-major f32 weight w (N x Kd), B[k][n]=w[n][k]
// dst layout: ((nt*KT + kt)*32 + lane)*16 + e
// ---------------------------------------------------------------------------
__global__ void k_prep_b(const float* __restrict__ src, _Float16* __restrict__ dst,
                         int Kd, int N) {
  int j = blockIdx.x * blockDim.x + threadIdx.x;
  if (j >= Kd * N) return;
  int e    = j & 15;
  int lane = (j >> 4) & 31;
  int tile = j >> 9;
  int KT   = Kd >> 5;
  int kt   = tile % KT;
  int nt   = tile / KT;
  int v    = e >> 1;
  int kb   = ((v & 4) << 2) + ((v & 3) << 1) + (e & 1);
  int k    = kt * 32 + ((lane >> 4) << 3) + kb;
  int n    = nt * 16 + (lane & 15);
  dst[j] = (_Float16)src[(size_t)n * Kd + k];
}

__global__ void k_cnorm(const float* __restrict__ cb, float* __restrict__ cn) {
  int t = blockIdx.x * blockDim.x + threadIdx.x;
  if (t >= 512) return;
  float s = 0.f;
  for (int d = 0; d < 128; ++d) { float v = cb[t * 128 + d]; s += v * v; }
  cn[t] = s;
}

__global__ void k_im2col(const float* __restrict__ img, _Float16* __restrict__ A) {
  int blk = blockIdx.x;                 // b*256 + n
  int b = blk >> 8, n = blk & 255;
  int gy = n >> 4, gx = n & 15;
  const float* base = img + (size_t)b * 3 * 65536;
  _Float16* dst = A + (size_t)blk * 768;
  for (int k = threadIdx.x; k < 768; k += 256) {
    int c = k >> 8, rem = k & 255, py = rem >> 4, px = rem & 15;
    dst[k] = (_Float16)base[(size_t)c * 65536 + (gy * 16 + py) * 256 + gx * 16 + px];
  }
}

// ---------------------------------------------------------------------------
// Stats encoder: single pass raw moments -> 15 features, then tiny MLP
// ---------------------------------------------------------------------------
__device__ __forceinline__ float bsum(float v, float* sh) {
  int t = threadIdx.x;
  sh[t] = v; __syncthreads();
  for (int s = blockDim.x >> 1; s; s >>= 1) { if (t < s) sh[t] += sh[t + s]; __syncthreads(); }
  float r = sh[0]; __syncthreads(); return r;
}
__device__ __forceinline__ float bmin(float v, float* sh) {
  int t = threadIdx.x;
  sh[t] = v; __syncthreads();
  for (int s = blockDim.x >> 1; s; s >>= 1) { if (t < s) sh[t] = fminf(sh[t], sh[t + s]); __syncthreads(); }
  float r = sh[0]; __syncthreads(); return r;
}
__device__ __forceinline__ float bmax(float v, float* sh) {
  int t = threadIdx.x;
  sh[t] = v; __syncthreads();
  for (int s = blockDim.x >> 1; s; s >>= 1) { if (t < s) sh[t] = fmaxf(sh[t], sh[t + s]); __syncthreads(); }
  float r = sh[0]; __syncthreads(); return r;
}

__global__ void k_stats(const float* __restrict__ img, float* __restrict__ feats) {
  __shared__ float sh[256];
  int b = blockIdx.x, t = threadIdx.x;
  const float* p = img + (size_t)b * 3 * 65536;
  float s0 = 0, s1 = 0, s2 = 0, q0 = 0, q1 = 0, q2 = 0;
  float mn0 = 1e30f, mn1 = 1e30f, mn2 = 1e30f, mx0 = -1e30f, mx1 = -1e30f, mx2 = -1e30f;
  float l1 = 0, l2 = 0, l3 = 0, l4 = 0, dk = 0;
  for (int i = t; i < 65536; i += 256) {
    float c0 = p[i], c1 = p[65536 + i], c2 = p[131072 + i];
    s0 += c0; s1 += c1; s2 += c2;
    q0 += c0 * c0; q1 += c1 * c1; q2 += c2 * c2;
    mn0 = fminf(mn0, c0); mn1 = fminf(mn1, c1); mn2 = fminf(mn2, c2);
    mx0 = fmaxf(mx0, c0); mx1 = fmaxf(mx1, c1); mx2 = fmaxf(mx2, c2);
    float l = 0.299f * c0 + 0.587f * c1 + 0.114f * c2;
    float ll = l * l;
    l1 += l; l2 += ll; l3 += ll * l; l4 += ll * ll;
    dk += (l < 0.2f) ? 1.f : 0.f;
  }
  s0 = bsum(s0, sh); s1 = bsum(s1, sh); s2 = bsum(s2, sh);
  q0 = bsum(q0, sh); q1 = bsum(q1, sh); q2 = bsum(q2, sh);
  mn0 = bmin(mn0, sh); mn1 = bmin(mn1, sh); mn2 = bmin(mn2, sh);
  mx0 = bmax(mx0, sh); mx1 = bmax(mx1, sh); mx2 = bmax(mx2, sh);
  l1 = bsum(l1, sh); l2 = bsum(l2, sh); l3 = bsum(l3, sh); l4 = bsum(l4, sh);
  dk = bsum(dk, sh);
  if (t == 0) {
    const float inv = 1.f / 65536.f;
    float* f = feats + b * 15;
    f[0] = s0 * inv; f[1] = s1 * inv; f[2] = s2 * inv;
    f[3] = sqrtf(fmaxf(0.f, q0 - s0 * s0 * inv) / 65535.f);
    f[4] = sqrtf(fmaxf(0.f, q1 - s1 * s1 * inv) / 65535.f);
    f[5] = sqrtf(fmaxf(0.f, q2 - s2 * s2 * inv) / 65535.f);
    f[6] = mn0; f[7] = mn1; f[8] = mn2;
    f[9] = mx0; f[10] = mx1; f[11] = mx2;
    float lm = l1 * inv, e2 = l2 * inv, e3 = l3 * inv, e4 = l4 * inv;
    float var = fmaxf(0.f, e2 - lm * lm);
    float sd = sqrtf(var);
    float m3 = e3 - 3.f * lm * e2 + 2.f * lm * lm * lm;
    float m4 = e4 - 4.f * lm * e3 + 6.f * lm * lm * e2 - 3.f * lm * lm * lm * lm;
    f[12] = m3 / (sd * sd * sd + 1e-6f);
    f[13] = m4 / (var * var + 1e-6f);
    f[14] = dk * inv;
  }
}

__global__ void k_statmlp(const float* __restrict__ feats, const float* __restrict__ w1,
                          const float* __restrict__ b1, const float* __restrict__ w2,
                          const float* __restrict__ b2, float* __restrict__ sf) {
  __shared__ float hid[128];
  __shared__ float f[15];
  int b = blockIdx.x, t = threadIdx.x;
  if (t < 15) f[t] = feats[b * 15 + t];
  __syncthreads();
  float a = b1[t];
#pragma unroll
  for (int j = 0; j < 15; ++j) a += f[j] * w1[t * 15 + j];
  hid[t] = fmaxf(a, 0.f);
  __syncthreads();
  float o = b2[t];
  for (int j = 0; j < 128; ++j) o += hid[j] * w2[t * 128 + j];
  sf[b * 128 + t] = o;
}

// ---------------------------------------------------------------------------
// LayerNorm (f32 in -> f16 out) and x->f16 + |z|^2 for VQ
// ---------------------------------------------------------------------------
__global__ void k_ln(const float* __restrict__ X, const float* __restrict__ g,
                     const float* __restrict__ be, _Float16* __restrict__ H) {
  __shared__ float sh[128];
  int row = blockIdx.x, t = threadIdx.x;
  float v = X[(size_t)row * 128 + t];
  sh[t] = v; __syncthreads();
  for (int s = 64; s; s >>= 1) { if (t < s) sh[t] += sh[t + s]; __syncthreads(); }
  float mu = sh[0] * (1.f / 128.f); __syncthreads();
  float d = v - mu;
  sh[t] = d * d; __syncthreads();
  for (int s = 64; s; s >>= 1) { if (t < s) sh[t] += sh[t + s]; __syncthreads(); }
  float r = rsqrtf(sh[0] * (1.f / 128.f) + 1e-5f);
  H[(size_t)row * 128 + t] = (_Float16)(d * r * g[t] + be[t]);
}

__global__ void k_znorm(const float* __restrict__ X, _Float16* __restrict__ XH,
                        float* __restrict__ zn) {
  __shared__ float sh[128];
  int row = blockIdx.x, t = threadIdx.x;
  float v = X[(size_t)row * 128 + t];
  XH[(size_t)row * 128 + t] = (_Float16)v;
  sh[t] = v * v; __syncthreads();
  for (int s = 64; s; s >>= 1) { if (t < s) sh[t] += sh[t + s]; __syncthreads(); }
  if (t == 0) zn[row] = sh[0];
}

// ---------------------------------------------------------------------------
// WMMA GEMM kernels with fused epilogues. blockDim=256 (8 waves), 1 tile/wave.
// ---------------------------------------------------------------------------
__global__ void __launch_bounds__(256) k_gemm_patch(const _Float16* __restrict__ A,
    const _Float16* __restrict__ W, const float* __restrict__ bias,
    const float* __restrict__ pos, const float* __restrict__ sf, float* __restrict__ X) {
  int tile = blockIdx.x * 8 + (threadIdx.x >> 5);
  int row0 = (tile >> 3) << 4, nt = tile & 7;   // tilesN = 128/16
  v8f acc = gemm_tile(A, W, 768, row0, nt);
  int lane = threadIdx.x & 31, mrel = (lane >> 4) << 3, nl = lane & 15;
  int d = (nt << 4) + nl;
#pragma unroll
  for (int r = 0; r < 8; ++r) {
    int m = row0 + r + mrel;
    int b = m >> 8, np = m & 255;
    X[(size_t)m * 128 + d] = acc[r] + bias[d] + pos[np * 128 + d] + sf[b * 128 + d];
  }
}

__global__ void __launch_bounds__(256) k_gemm_biasf16(const _Float16* __restrict__ A,
    const _Float16* __restrict__ W, int Kd, int N, const float* __restrict__ bias,
    _Float16* __restrict__ Out) {
  int tile = blockIdx.x * 8 + (threadIdx.x >> 5);
  int tilesN = N >> 4;
  int row0 = (tile / tilesN) << 4, nt = tile % tilesN;
  v8f acc = gemm_tile(A, W, Kd, row0, nt);
  int lane = threadIdx.x & 31, mrel = (lane >> 4) << 3, nl = lane & 15;
  int c = (nt << 4) + nl;
#pragma unroll
  for (int r = 0; r < 8; ++r) {
    int m = row0 + r + mrel;
    Out[(size_t)m * N + c] = (_Float16)(acc[r] + bias[c]);
  }
}

__global__ void __launch_bounds__(256) k_gemm_gelu(const _Float16* __restrict__ A,
    const _Float16* __restrict__ W, int Kd, int N, const float* __restrict__ bias,
    _Float16* __restrict__ Out) {
  int tile = blockIdx.x * 8 + (threadIdx.x >> 5);
  int tilesN = N >> 4;
  int row0 = (tile / tilesN) << 4, nt = tile % tilesN;
  v8f acc = gemm_tile(A, W, Kd, row0, nt);
  int lane = threadIdx.x & 31, mrel = (lane >> 4) << 3, nl = lane & 15;
  int c = (nt << 4) + nl;
#pragma unroll
  for (int r = 0; r < 8; ++r) {
    int m = row0 + r + mrel;
    float t = acc[r] + bias[c];
    Out[(size_t)m * N + c] = (_Float16)(0.5f * t * (1.f + erff(t * 0.70710678118654752f)));
  }
}

// residual: X = X + A*W + bias   (N fixed at 128)
__global__ void __launch_bounds__(256) k_gemm_res(const _Float16* __restrict__ A,
    const _Float16* __restrict__ W, int Kd, const float* __restrict__ bias,
    float* __restrict__ X) {
  int tile = blockIdx.x * 8 + (threadIdx.x >> 5);
  int row0 = (tile >> 3) << 4, nt = tile & 7;
  v8f acc = gemm_tile(A, W, Kd, row0, nt);
  int lane = threadIdx.x & 31, mrel = (lane >> 4) << 3, nl = lane & 15;
  int c = (nt << 4) + nl;
#pragma unroll
  for (int r = 0; r < 8; ++r) {
    int m = row0 + r + mrel;
    X[(size_t)m * 128 + c] = X[(size_t)m * 128 + c] + acc[r] + bias[c];
  }
}

// VQ: dist = |z|^2 + |c|^2 - 2 z.c ; fused 16-lane argmin + u64 atomicMin
__global__ void __launch_bounds__(256) k_gemm_vq(const _Float16* __restrict__ A,
    const _Float16* __restrict__ W, const float* __restrict__ zn,
    const float* __restrict__ cn, unsigned long long* __restrict__ mk) {
  int tile = blockIdx.x * 8 + (threadIdx.x >> 5);
  int row0 = (tile >> 5) << 4, nt = tile & 31;   // tilesN = 512/16 = 32
  v8f acc = gemm_tile(A, W, 128, row0, nt);
  int lane = threadIdx.x & 31, mrel = (lane >> 4) << 3, nl = lane & 15;
  int c = (nt << 4) + nl;
  float cnv = cn[c];
#pragma unroll
  for (int r = 0; r < 8; ++r) {
    int m = row0 + r + mrel;
    float db = zn[m] + cnv - 2.f * acc[r];
    int cb = c;
#pragma unroll
    for (int off = 1; off < 16; off <<= 1) {
      float od = __shfl_xor(db, off, 16);
      int   oc = __shfl_xor(cb, off, 16);
      if (od < db || (od == db && oc < cb)) { db = od; cb = oc; }
    }
    if (nl == 0) {
      unsigned u = __float_as_uint(db);
      u = (u & 0x80000000u) ? ~u : (u | 0x80000000u);  // order-preserving map
      atomicMin(&mk[m], ((unsigned long long)u << 32) | (unsigned)cb);
    }
  }
}

// ---------------------------------------------------------------------------
// Fused attention per (batch, head): K/V + per-wave P in LDS, all-WMMA math.
// V is staged in LDS already swizzled into B-fragment layout (16 tiles).
// ---------------------------------------------------------------------------
__global__ void __launch_bounds__(256) k_attn(const _Float16* __restrict__ qkv,
                                              _Float16* __restrict__ o) {
  extern __shared__ _Float16 sm[];
  _Float16* Kh = sm;            // [256][32] row-major
  _Float16* Vh = sm + 8192;     // swizzled: ((tv*8+kt)*32+lane)*16+e
  _Float16* Pb = sm + 16384;    // [8 waves][16][256]
  int b = blockIdx.x >> 2, hh = blockIdx.x & 3;
  const _Float16* base = qkv + (size_t)b * 256 * 384;
  for (int i = threadIdx.x; i < 256 * 32; i += 256) {
    int nrow = i >> 5, d = i & 31;
    Kh[i] = base[nrow * 384 + 128 + hh * 32 + d];
    // V swizzle: dest j=i -> (tile, lane, e) -> source (k=seq row, n=head col)
    int e = i & 15, lane = (i >> 4) & 31, tile = i >> 9;
    int tv = tile >> 3, kt = tile & 7;
    int v = e >> 1;
    int kb = ((v & 4) << 2) + ((v & 3) << 1) + (e & 1);
    int k = kt * 32 + ((lane >> 4) << 3) + kb;
    int n = tv * 16 + (lane & 15);
    Vh[i] = base[k * 384 + 256 + hh * 32 + n];
  }
  __syncthreads();
  int wave = threadIdx.x >> 5, lane = threadIdx.x & 31;
  int mrel = (lane >> 4) << 3, nl = lane & 15;
  _Float16* P = Pb + wave * 16 * 256;
  const float scale = 0.17677669529663687f;   // 1/sqrt(32)
  for (int qi = 0; qi < 2; ++qi) {
    int row0 = (wave + qi * 8) << 4;
    v16h aq = frag_a(base + hh * 32, 384, row0, 0);   // q tile 16x32
    v8f s[16];
#pragma unroll
    for (int t = 0; t < 16; ++t) {
      v16h bk = frag_a(Kh, 32, t * 16, 0);            // K^T column tile == K rows
      v8f z = {0.f, 0.f, 0.f, 0.f, 0.f, 0.f, 0.f, 0.f};
      s[t] = wmma32(aq, bk, z);
    }
#pragma unroll
    for (int r = 0; r < 8; ++r) {
      float mx = -1e30f;
#pragma unroll
      for (int t = 0; t < 16; ++t) mx = fmaxf(mx, s[t][r] * scale);
#pragma unroll
      for (int off = 1; off < 16; off <<= 1) mx = fmaxf(mx, __shfl_xor(mx, off, 16));
      float sum = 0.f;
#pragma unroll
      for (int t = 0; t < 16; ++t) { float e = __expf(s[t][r] * scale - mx); s[t][r] = e; sum += e; }
#pragma unroll
      for (int off = 1; off < 16; off <<= 1) sum += __shfl_xor(sum, off, 16);
      float inv = 1.f / sum;
      int m = r + mrel;
#pragma unroll
      for (int t = 0; t < 16; ++t) P[m * 256 + t * 16 + nl] = (_Float16)(s[t][r] * inv);
    }
    // O = P(16x256) * V(256x32), V already fragment-swizzled in LDS
#pragma unroll
    for (int tv = 0; tv < 2; ++tv) {
      v8f acc = {0.f, 0.f, 0.f, 0.f, 0.f, 0.f, 0.f, 0.f};
#pragma unroll
      for (int kt = 0; kt < 8; ++kt)
        acc = wmma32(frag_a(P, 256, 0, kt * 32), frag_b_sw(Vh, 8, kt, tv), acc);
#pragma unroll
      for (int r = 0; r < 8; ++r) {
        int m = row0 + r + mrel;
        o[((size_t)b * 256 + m) * 128 + hh * 32 + tv * 16 + nl] = (_Float16)acc[r];
      }
    }
  }
}

// ---------------------------------------------------------------------------
// VQ finalize / outputs
// ---------------------------------------------------------------------------
__global__ void k_init(unsigned long long* mk, unsigned* counts) {
  int t = blockIdx.x * blockDim.x + threadIdx.x;
  if (t < 16384) mk[t] = 0xFFFFFFFFFFFFFFFFull;
  if (t < 512) counts[t] = 0u;
}

__global__ void k_vq_out(const unsigned long long* __restrict__ mk,
                         const float* __restrict__ cb, float* __restrict__ quant,
                         int* __restrict__ idxout, float* __restrict__ gf,
                         unsigned* __restrict__ counts) {
  __shared__ int sidx[256];
  int b = blockIdx.x, t = threadIdx.x;   // 128 threads
  for (int n = t; n < 256; n += 128) {
    int id = (int)(mk[b * 256 + n] & 0xFFFFFFFFull);
    sidx[n] = id;
    idxout[b * 256 + n] = id;
    atomicAdd(&counts[id], 1u);
  }
  __syncthreads();
  float sum = 0.f;
  for (int n = 0; n < 256; ++n) {
    float v = cb[(size_t)sidx[n] * 128 + t];
    quant[((size_t)b * 256 + n) * 128 + t] = v;   // quant value == codebook[idx]
    sum += v;
  }
  gf[b * 128 + t] = sum * (1.f / 256.f);
}

__global__ void k_logits(const float* __restrict__ gf, const float* __restrict__ hw,
                         const float* __restrict__ hb, float* __restrict__ out) {
  int t = threadIdx.x;       // 512 = 64 * 8
  int b = t >> 3, c = t & 7;
  float s = hb[c];
  for (int d = 0; d < 128; ++d) s += gf[b * 128 + d] * hw[c * 128 + d];
  out[t] = s;
}

__global__ void k_perp(const unsigned* __restrict__ counts, float* __restrict__ out) {
  __shared__ float sh[512];
  int t = threadIdx.x;
  float pr = counts[t] * (1.f / 16384.f);
  sh[t] = pr * logf(pr + 1e-10f);
  __syncthreads();
  for (int s = 256; s; s >>= 1) { if (t < s) sh[t] += sh[t + s]; __syncthreads(); }
  if (t == 0) out[0] = expf(-sh[0]);
}

// ---------------------------------------------------------------------------
extern "C" void kernel_launch(void* const* d_in, const int* in_sizes, int n_in,
                              void* d_out, int out_size, void* d_ws, size_t ws_size,
                              hipStream_t stream) {
  const float* image   = (const float*)d_in[0];
  const float* conv_w  = (const float*)d_in[1];
  const float* conv_b  = (const float*)d_in[2];
  const float* stat_w1 = (const float*)d_in[3];
  const float* stat_b1 = (const float*)d_in[4];
  const float* stat_w2 = (const float*)d_in[5];
  const float* stat_b2 = (const float*)d_in[6];
  const float* pos     = (const float*)d_in[7];
  const float* ln1_s   = (const float*)d_in[8];
  const float* ln1_b   = (const float*)d_in[9];
  const float* qkv_w   = (const float*)d_in[10];
  const float* qkv_b   = (const float*)d_in[11];
  const float* out_w   = (const float*)d_in[12];
  const float* out_b   = (const float*)d_in[13];
  const float* ln2_s   = (const float*)d_in[14];
  const float* ln2_b   = (const float*)d_in[15];
  const float* ff1_w   = (const float*)d_in[16];
  const float* ff1_b   = (const float*)d_in[17];
  const float* ff2_w   = (const float*)d_in[18];
  const float* ff2_b   = (const float*)d_in[19];
  const float* cbook   = (const float*)d_in[20];
  const float* head_w  = (const float*)d_in[21];
  const float* head_b  = (const float*)d_in[22];

  // ---- workspace layout (bytes) ----
  char* W = (char*)d_ws;
  _Float16* Wc    = (_Float16*)(W + 0);         // 768x128 swizzled
  _Float16* Wqkv  = (_Float16*)(W + 196608);    // 2 x 128x384 swizzled
  _Float16* Wout  = (_Float16*)(W + 393216);    // 2 x 128x128 swizzled
  _Float16* Wff1  = (_Float16*)(W + 458752);    // 2 x 128x256 swizzled
  _Float16* Wff2  = (_Float16*)(W + 589824);    // 2 x 256x128 swizzled
  _Float16* Wcb   = (_Float16*)(W + 720896);    // 128x512 swizzled
  float*    cnorm = (float*)(W + 851968);
  float*    feats = (float*)(W + 854016);
  float*    sf    = (float*)(W + 858112);
  float*    znorm = (float*)(W + 890880);
  unsigned long long* minkey = (unsigned long long*)(W + 956416);
  unsigned* counts = (unsigned*)(W + 1087488);
  float*    X     = (float*)(W + 2097152);          // 16384x128 f32
  _Float16* H     = (_Float16*)(W + 10485760);      // 16384x128 f16 (also xh)
  _Float16* Amat  = (_Float16*)(W + 14680064);      // 16384x768 f16
  _Float16* QKV   = Amat;                           // reuse: 16384x384 f16
  _Float16* Obuf  = (_Float16*)(W + 14680064 + 12582912);  // 16384x128 f16
  _Float16* G     = (_Float16*)(W + 14680064 + 16777216);  // 16384x256 f16

  // ---- output layout ----
  float* out    = (float*)d_out;
  float* quant  = out;                       // 64*256*128
  int*   idxout = (int*)(out + 2097152);     // 64*256
  float* logits = out + 2097152 + 16384;     // 64*8
  float* gfout  = logits + 512;              // 64*128
  float* perp   = gfout + 8192;              // 1

  auto PB = [&](const float* s, _Float16* d, int Kd, int N) {
    k_prep_b<<<(Kd * N + 255) / 256, 256, 0, stream>>>(s, d, Kd, N);
  };
  PB(conv_w,           Wc,              768, 128);
  PB(qkv_w,            Wqkv,            128, 384);
  PB(qkv_w + 384*128,  Wqkv + 128*384,  128, 384);
  PB(out_w,            Wout,            128, 128);
  PB(out_w + 128*128,  Wout + 128*128,  128, 128);
  PB(ff1_w,            Wff1,            128, 256);
  PB(ff1_w + 256*128,  Wff1 + 128*256,  128, 256);
  PB(ff2_w,            Wff2,            256, 128);
  PB(ff2_w + 128*256,  Wff2 + 256*128,  256, 128);
  PB(cbook,            Wcb,             128, 512);
  k_cnorm<<<2, 256, 0, stream>>>(cbook, cnorm);

  k_stats<<<64, 256, 0, stream>>>(image, feats);
  k_statmlp<<<64, 128, 0, stream>>>(feats, stat_w1, stat_b1, stat_w2, stat_b2, sf);

  k_im2col<<<16384, 256, 0, stream>>>(image, Amat);
  k_gemm_patch<<<1024, 256, 0, stream>>>(Amat, Wc, conv_b, pos, sf, X);

  for (int i = 0; i < 2; ++i) {
    k_ln<<<16384, 128, 0, stream>>>(X, ln1_s + i * 128, ln1_b + i * 128, H);
    k_gemm_biasf16<<<3072, 256, 0, stream>>>(H, Wqkv + (size_t)i * 128 * 384, 128, 384,
                                             qkv_b + i * 384, QKV);
    k_attn<<<256, 256, 98304, stream>>>(QKV, Obuf);
    k_gemm_res<<<1024, 256, 0, stream>>>(Obuf, Wout + (size_t)i * 128 * 128, 128,
                                         out_b + i * 128, X);
    k_ln<<<16384, 128, 0, stream>>>(X, ln2_s + i * 128, ln2_b + i * 128, H);
    k_gemm_gelu<<<2048, 256, 0, stream>>>(H, Wff1 + (size_t)i * 128 * 256, 128, 256,
                                          ff1_b + i * 256, G);
    k_gemm_res<<<1024, 256, 0, stream>>>(G, Wff2 + (size_t)i * 256 * 128, 256,
                                         ff2_b + i * 128, X);
  }

  k_znorm<<<16384, 128, 0, stream>>>(X, H, znorm);
  k_init<<<64, 256, 0, stream>>>(minkey, counts);
  k_gemm_vq<<<4096, 256, 0, stream>>>(H, Wcb, znorm, cnorm, minkey);
  k_vq_out<<<64, 128, 0, stream>>>(minkey, cbook, quant, idxout, gfout, counts);
  k_logits<<<1, 512, 0, stream>>>(gfout, head_w, head_b, logits);
  k_perp<<<1, 512, 0, stream>>>(counts, perp);
}
